// EnhancedAttentiveFP_24086176596131
// MI455X (gfx1250) — compile-verified, matrix operand order checked by reading, not run on hardware
//
#include <hip/hip_runtime.h>
#include <hip/hip_bf16.h>
#include <cmath>

typedef __attribute__((ext_vector_type(16))) __bf16 v16bf;
typedef __attribute__((ext_vector_type(8)))  __bf16 v8bf;
typedef __attribute__((ext_vector_type(8)))  float  v8f;

#define HDIM 256
#define LEAKY(v) ((v) > 0.f ? (v) : 0.01f * (v))

// ---------------------------------------------------------------------------
// elementwise / utility kernels
// ---------------------------------------------------------------------------
__global__ void cast_bf16_kernel(const float* __restrict__ src, __bf16* __restrict__ dst, int n) {
  int i = blockIdx.x * blockDim.x + threadIdx.x;
  if (i < n) dst[i] = (__bf16)src[i];
}

__global__ void cast_bf16_strided_kernel(const float* __restrict__ src, int srcStride,
                                         __bf16* __restrict__ dst, int rows, int cols) {
  int i = blockIdx.x * blockDim.x + threadIdx.x;
  if (i < rows * cols) {
    int r = i / cols, c = i % cols;
    dst[i] = (__bf16)src[(size_t)r * srcStride + c];
  }
}

__global__ void fill_kernel(float* __restrict__ p, float v, int n) {
  int i = blockIdx.x * blockDim.x + threadIdx.x;
  if (i < n) p[i] = v;
}

__global__ void rows_bias_kernel(float* __restrict__ dst, const float* __restrict__ bias, int rows) {
  int i = blockIdx.x * blockDim.x + threadIdx.x;
  if (i < rows * HDIM) dst[i] = bias[i & (HDIM - 1)];
}

__global__ void elu_kernel(float* __restrict__ p, int n) {
  int i = blockIdx.x * blockDim.x + threadIdx.x;
  if (i < n) { float v = p[i]; p[i] = v > 0.f ? v : (__expf(v) - 1.f); }
}

__global__ void relu_kernel(float* __restrict__ p, int n) {
  int i = blockIdx.x * blockDim.x + threadIdx.x;
  if (i < n) p[i] = fmaxf(p[i], 0.f);
}

// ---------------------------------------------------------------------------
// WMMA bf16 GEMM: C[M,Nout] = act(A[M,K] @ W[Nout,K]^T + bias)
// A,W bf16 row-major; C f32. Requires M%16==0, K%32==0 (K<=256), Nout%64==0.
// Block: 256 threads = 8 waves. Wave w owns rows [by*256 + w*32, +32) as two
// 16-row tiles sharing four LDS-resident 16x16 B tiles (cols [bx*64,+64)).
// Per k-step: 4 global_load_b128 (A) + 8 ds_load_b128 (B) feed 8 v_wmma.
// ---------------------------------------------------------------------------
__global__ __launch_bounds__(256) void gemm_bf16_wmma_kernel(
    const __bf16* __restrict__ A, const __bf16* __restrict__ W,
    const float* __restrict__ bias, float* __restrict__ C,
    int M, int K, int Nout, int act) {
  __shared__ __align__(16) __bf16 ldsW[64 * 256];

  const int n0 = blockIdx.x * 64;
  // cooperative stage of W[n0 .. n0+64, 0..K)  (contiguous 64*K bf16)
  const __bf16* wsrc = W + (size_t)n0 * K;
  const int total = 64 * K;
  for (int i = threadIdx.x * 8; i < total; i += 256 * 8)
    *(v8bf*)&ldsW[i] = *(const v8bf*)&wsrc[i];
  __syncthreads();

  const int wave = threadIdx.x >> 5;
  const int lane = threadIdx.x & 31;
  const int lm = lane & 15;   // row-in-tile (A) / col (B,D)
  const int hs = lane >> 4;   // K half selector
  const int m0 = blockIdx.y * 256 + wave * 32;  // first 16-row tile
  const int m1 = m0 + 16;                       // second 16-row tile
  if (m0 < M) {
    const bool two = (m1 < M);                  // wave-uniform
    v8f acc[8] = {v8f{}, v8f{}, v8f{}, v8f{}, v8f{}, v8f{}, v8f{}, v8f{}};
    const __bf16* arow0 = A + (size_t)(m0 + lm) * K + hs * 8;
    const __bf16* arow1 = A + (size_t)(m1 + lm) * K + hs * 8;
    for (int k0 = 0; k0 < K; k0 += 32) {
      // A fragments (16-bit A 16x32 layout per ISA 7.12.2)
      v8bf a0lo = *(const v8bf*)(arow0 + k0);
      v8bf a0hi = *(const v8bf*)(arow0 + k0 + 16);
      v16bf af0;
#pragma unroll
      for (int e = 0; e < 8; ++e) { af0[e] = a0lo[e]; af0[8 + e] = a0hi[e]; }
      v16bf af1;
      if (two) {
        v8bf a1lo = *(const v8bf*)(arow1 + k0);
        v8bf a1hi = *(const v8bf*)(arow1 + k0 + 16);
#pragma unroll
        for (int e = 0; e < 8; ++e) { af1[e] = a1lo[e]; af1[8 + e] = a1hi[e]; }
      }
      // all four B fragments up front -> clause the ds_loads, wmma back-to-back
      v16bf bfr[4];
#pragma unroll
      for (int t = 0; t < 4; ++t) {
        const __bf16* wr = &ldsW[(size_t)(t * 16 + lm) * K + k0 + hs * 8];
        v8bf blo = *(const v8bf*)wr;
        v8bf bhi = *(const v8bf*)(wr + 16);
#pragma unroll
        for (int e = 0; e < 8; ++e) { bfr[t][e] = blo[e]; bfr[t][8 + e] = bhi[e]; }
      }
#pragma unroll
      for (int t = 0; t < 4; ++t)
        acc[t] = __builtin_amdgcn_wmma_f32_16x16x32_bf16(
            false, af0, false, bfr[t], (short)0, acc[t], false, false);
      if (two) {
#pragma unroll
        for (int t = 0; t < 4; ++t)
          acc[4 + t] = __builtin_amdgcn_wmma_f32_16x16x32_bf16(
              false, af1, false, bfr[t], (short)0, acc[4 + t], false, false);
      }
    }
    // store (C/D layout: VGPR v -> m = v + 8*hs, lane%16 -> n)
#pragma unroll
    for (int t = 0; t < 4; ++t) {
      int n = n0 + t * 16 + lm;
      float b = bias ? bias[n] : 0.f;
#pragma unroll
      for (int v = 0; v < 8; ++v) {
        int m = m0 + hs * 8 + v;
        float val = acc[t][v] + b;
        if (act == 1) val = LEAKY(val);
        else if (act == 2) val = fmaxf(val, 0.f);
        C[(size_t)m * Nout + n] = val;
      }
    }
    if (two) {
#pragma unroll
      for (int t = 0; t < 4; ++t) {
        int n = n0 + t * 16 + lm;
        float b = bias ? bias[n] : 0.f;
#pragma unroll
        for (int v = 0; v < 8; ++v) {
          int m = m1 + hs * 8 + v;
          float val = acc[4 + t][v] + b;
          if (act == 1) val = LEAKY(val);
          else if (act == 2) val = fmaxf(val, 0.f);
          C[(size_t)m * Nout + n] = val;
        }
      }
    }
  }
}

// ---------------------------------------------------------------------------
// wave reductions
// ---------------------------------------------------------------------------
__device__ inline float wave_sum(float v) {
#pragma unroll
  for (int off = 16; off > 0; off >>= 1) v += __shfl_xor(v, off, 32);
  return v;
}

// ---------------------------------------------------------------------------
// GATEConv attention logit: one wave per edge.
// alpha[e] = leaky( sum_c leaky(px[row,c] + (edge_attr[e] @ We^T)[c]) * att_l[c]
//                   + xr[col] )
// ---------------------------------------------------------------------------
__global__ __launch_bounds__(256) void gate_alpha_kernel(
    const float* __restrict__ px, const float* __restrict__ eattr,
    const float* __restrict__ Wg /*[256,272]*/, const float* __restrict__ att_l,
    const float* __restrict__ xr, const int* __restrict__ row,
    const int* __restrict__ col, float* __restrict__ alpha, int E) {
  int e = blockIdx.x * 8 + (threadIdx.x >> 5);
  int lane = threadIdx.x & 31;
  if (e >= E) return;
  int r = row[e], c = col[e];
  float ea[16];
#pragma unroll
  for (int f = 0; f < 16; ++f) ea[f] = eattr[(size_t)e * 16 + f];
  float partial = 0.f;
#pragma unroll
  for (int j = 0; j < 8; ++j) {
    int ch = lane + 32 * j;
    float pe = 0.f;
#pragma unroll
    for (int f = 0; f < 16; ++f) pe = fmaf(ea[f], Wg[(size_t)ch * 272 + 256 + f], pe);
    float hj = px[(size_t)r * HDIM + ch] + pe;
    hj = LEAKY(hj);
    partial = fmaf(hj, att_l[ch], partial);
  }
  partial = wave_sum(partial);
  if (lane == 0) { float v = partial + xr[c]; alpha[e] = LEAKY(v); }
}

// y[r] = dot(X[r,0:256], w)  — one wave per row
__global__ __launch_bounds__(256) void rowdot_kernel(
    const float* __restrict__ X, const float* __restrict__ w,
    float* __restrict__ out, int rows) {
  int r = blockIdx.x * 8 + (threadIdx.x >> 5);
  int lane = threadIdx.x & 31;
  if (r >= rows) return;
  float s = 0.f;
#pragma unroll
  for (int j = 0; j < 8; ++j) { int c = lane + 32 * j; s = fmaf(X[(size_t)r * HDIM + c], w[c], s); }
  s = wave_sum(s);
  if (lane == 0) out[r] = s;
}

__global__ void gat_alpha_kernel(const float* __restrict__ vs, const float* __restrict__ vd,
                                 const int* __restrict__ row, const int* __restrict__ col,
                                 float* __restrict__ al, int E) {
  int e = blockIdx.x * blockDim.x + threadIdx.x;
  if (e < E) { float v = vs[row[e]] + vd[col[e]]; al[e] = LEAKY(v); }
}

__global__ void mol_alpha_kernel(const float* __restrict__ asrc, const float* __restrict__ ad,
                                 const int* __restrict__ batch, float* __restrict__ al, int n) {
  int i = blockIdx.x * blockDim.x + threadIdx.x;
  if (i < n) { float v = asrc[i] + ad[batch[i]]; al[i] = LEAKY(v); }
}

// ---------------------------------------------------------------------------
// segment softmax pieces (atomics)
// ---------------------------------------------------------------------------
__device__ inline void atomicMaxF(float* addr, float val) {
  if (val >= 0.f) atomicMax((int*)addr, __float_as_int(val));
  else            atomicMin((unsigned int*)addr, __float_as_uint(val));
}

__global__ void seg_max_kernel(const float* __restrict__ v, const int* __restrict__ seg,
                               float* __restrict__ m, int n) {
  int i = blockIdx.x * blockDim.x + threadIdx.x;
  if (i < n) atomicMaxF(&m[seg[i]], v[i]);
}

__global__ void seg_exp_kernel(const float* __restrict__ alpha, const int* __restrict__ seg,
                               const float* __restrict__ m, float* __restrict__ e,
                               float* __restrict__ s, int n) {
  int i = blockIdx.x * blockDim.x + threadIdx.x;
  if (i >= n) return;
  float mm = m[seg[i]];
  if (mm < -1e37f) mm = 0.f;            // empty-segment -inf -> 0 (matches reference)
  float ex = __expf(alpha[i] - mm);
  e[i] = ex;
  atomicAdd(&s[seg[i]], ex);
}

// out[dst[i], :] += msg[src?src[i]:i, :] * (e?e[i]/(s[dst]+1e-16):1)  — wave/item
__global__ __launch_bounds__(256) void scatter_wsum_kernel(
    const float* __restrict__ msg, const float* __restrict__ e,
    const float* __restrict__ s, const int* __restrict__ src,
    const int* __restrict__ dst, float* __restrict__ out, int n) {
  int i = blockIdx.x * 8 + (threadIdx.x >> 5);
  int lane = threadIdx.x & 31;
  if (i >= n) return;
  int si = src ? src[i] : i;
  int d = dst[i];
  float a = 1.f;
  if (e) a = e[i] / (s[d] + 1e-16f);
  const float* mrow = msg + (size_t)si * HDIM;
  float* orow = out + (size_t)d * HDIM;
#pragma unroll
  for (int j = 0; j < 8; ++j) { int c = lane + 32 * j; atomicAdd(&orow[c], mrow[c] * a); }
}

// ---------------------------------------------------------------------------
// GRU elementwise combine (gi/gh are [rows,768]); writes h state in place
// ---------------------------------------------------------------------------
__global__ void gru_combine_kernel(const float* __restrict__ gi, const float* __restrict__ gh,
                                   float* __restrict__ hx, int rows) {
  int i = blockIdx.x * blockDim.x + threadIdx.x;
  if (i >= rows * HDIM) return;
  int r = i >> 8, c = i & 255;
  size_t b = (size_t)r * 768;
  float ir = gi[b + c], iz = gi[b + 256 + c], in_ = gi[b + 512 + c];
  float hr = gh[b + c], hz = gh[b + 256 + c], hn = gh[b + 512 + c];
  float rg = 1.f / (1.f + __expf(-(ir + hr)));
  float z  = 1.f / (1.f + __expf(-(iz + hz)));
  float nn = tanhf(in_ + rg * hn);
  float hp = hx[i];
  float o = (1.f - z) * nn + z * hp;
  hx[i] = fmaxf(o, 0.f);                 // every GRU in reference is relu-wrapped
}

// ---------------------------------------------------------------------------
// head: LayerNorm -> ReLU -> dot(out_W) + out_b  — one wave per graph
// ---------------------------------------------------------------------------
__global__ __launch_bounds__(256) void head_kernel(
    const float* __restrict__ z, const float* __restrict__ ln_g,
    const float* __restrict__ ln_b, const float* __restrict__ outW,
    const float* __restrict__ outb, float* __restrict__ out, int Gn) {
  int g = blockIdx.x * 8 + (threadIdx.x >> 5);
  int lane = threadIdx.x & 31;
  if (g >= Gn) return;
  float v[8], s = 0.f, s2 = 0.f;
#pragma unroll
  for (int j = 0; j < 8; ++j) {
    v[j] = z[(size_t)g * HDIM + lane + 32 * j];
    s += v[j]; s2 = fmaf(v[j], v[j], s2);
  }
  s = wave_sum(s); s2 = wave_sum(s2);
  float mu = s * (1.f / 256.f);
  float var = s2 * (1.f / 256.f) - mu * mu;
  float inv = rsqrtf(var + 1e-5f);
  float acc = 0.f;
#pragma unroll
  for (int j = 0; j < 8; ++j) {
    int c = lane + 32 * j;
    float zn = (v[j] - mu) * inv * ln_g[c] + ln_b[c];
    zn = fmaxf(zn, 0.f);
    acc = fmaf(zn, outW[c], acc);
  }
  acc = wave_sum(acc);
  if (lane == 0) out[g] = acc + outb[0];
}

// ---------------------------------------------------------------------------
// host orchestration
// ---------------------------------------------------------------------------
extern "C" void kernel_launch(void* const* d_in, const int* in_sizes, int n_in,
                              void* d_out, int out_size, void* d_ws, size_t ws_size,
                              hipStream_t stream) {
  (void)n_in; (void)ws_size;
  const int N = in_sizes[3];       // batch has N entries
  const int E = in_sizes[1] / 2;   // edge_index [2,E]
  const int G = out_size;          // one scalar per graph

  const float* x_in  = (const float*)d_in[0];
  const int*   ei    = (const int*)d_in[1];
  const float* eattr = (const float*)d_in[2];
  const int*   batch = (const int*)d_in[3];
  const int* row = ei;
  const int* col = ei + E;

  auto P = [&](int i) -> const float* { return (const float*)d_in[i]; };
  // params flattened in setup_inputs() insertion order:
  const float* lin1_W = P(4);  const float* lin1_b = P(5);
  const float* g_W1 = P(6);    const float* g_W2 = P(7);
  const float* g_attl = P(8);  const float* g_attr = P(9);  const float* g_bias = P(10);
  // gru dicts: Wih, Whh, bih, bhh. gru0 @11; atom_grus @27+4L; mol_gru @43
  // gat dicts: W, att_src, att_dst, bias. atom_convs @15+4L; mol_conv @39
  const float* lin2_W = P(47); const float* lin2_b = P(48);
  const float* head_W = P(49); const float* head_b = P(50);
  const float* ln_g = P(51);   const float* ln_b = P(52);
  const float* out_W = P(53);  const float* out_b = P(54);

  // ---- workspace bump allocator ----
  char* wp = (char*)d_ws;
  auto alloc = [&](size_t bytes) -> void* {
    void* p = (void*)wp; wp += (bytes + 255) & ~(size_t)255; return p;
  };
  float* X    = (float*)alloc((size_t)N * 256 * 4);
  float* PX   = (float*)alloc((size_t)N * 256 * 4);
  float* HN   = (float*)alloc((size_t)N * 256 * 4);
  float* GI   = (float*)alloc((size_t)N * 768 * 4);
  float* GH   = (float*)alloc((size_t)N * 768 * 4);
  float* AL   = (float*)alloc((size_t)E * 4);
  float* EE   = (float*)alloc((size_t)E * 4);
  float* SM   = (float*)alloc((size_t)N * 4);
  float* SS   = (float*)alloc((size_t)N * 4);
  float* VR   = (float*)alloc((size_t)N * 4);
  float* VR2  = (float*)alloc((size_t)N * 4);
  float* AD   = (float*)alloc((size_t)G * 4);
  float* OUTG = (float*)alloc((size_t)G * 256 * 4);
  float* HG   = (float*)alloc((size_t)G * 256 * 4);
  float* GIG  = (float*)alloc((size_t)G * 768 * 4);
  float* GHG  = (float*)alloc((size_t)G * 768 * 4);
  __bf16* XB  = (__bf16*)alloc((size_t)N * 256 * 2);
  __bf16* XB2 = (__bf16*)alloc((size_t)N * 256 * 2);
  __bf16* OB  = (__bf16*)alloc((size_t)G * 256 * 2);
  __bf16* OB2 = (__bf16*)alloc((size_t)G * 256 * 2);
  __bf16* wb_lin1   = (__bf16*)alloc(256 * 64 * 2);
  __bf16* wb_gateWx = (__bf16*)alloc(256 * 256 * 2);
  __bf16* wb_gateW2 = (__bf16*)alloc(256 * 256 * 2);
  __bf16 *wb_Wih[5], *wb_Whh[5], *wb_gat[4];
  for (int i = 0; i < 5; ++i) { wb_Wih[i] = (__bf16*)alloc(768 * 256 * 2);
                                wb_Whh[i] = (__bf16*)alloc(768 * 256 * 2); }
  for (int i = 0; i < 4; ++i)   wb_gat[i] = (__bf16*)alloc(256 * 256 * 2);
  __bf16* wb_lin2 = (__bf16*)alloc(256 * 256 * 2);
  __bf16* wb_head = (__bf16*)alloc(256 * 256 * 2);

  // ---- launch helpers ----
  auto cgrid = [](int n) { return dim3((unsigned)((n + 255) / 256)); };
  auto wgrid = [](int n) { return dim3((unsigned)((n + 7) / 8)); };
  auto castb = [&](const float* s, __bf16* d, int n) {
    cast_bf16_kernel<<<cgrid(n), 256, 0, stream>>>(s, d, n);
  };
  auto gemm = [&](const __bf16* A, const __bf16* W, const float* bias, float* C,
                  int M, int K, int Nout, int act) {
    dim3 g((unsigned)(Nout / 64), (unsigned)((M + 255) / 256));
    gemm_bf16_wmma_kernel<<<g, 256, 0, stream>>>(A, W, bias, C, M, K, Nout, act);
  };
  auto softmax_seg = [&](float* a, const int* seg, int count, int nseg) {
    fill_kernel<<<cgrid(nseg), 256, 0, stream>>>(SM, -INFINITY, nseg);
    fill_kernel<<<cgrid(nseg), 256, 0, stream>>>(SS, 0.f, nseg);
    seg_max_kernel<<<cgrid(count), 256, 0, stream>>>(a, seg, SM, count);
    seg_exp_kernel<<<cgrid(count), 256, 0, stream>>>(a, seg, SM, EE, SS, count);
  };
  auto run_gru = [&](const float* h_in, float* h_state, int gidx, int rows,
                     __bf16* t1, __bf16* t2, float* gi, float* gh,
                     const float* bih, const float* bhh) {
    castb(h_in, t1, rows * 256);
    gemm(t1, wb_Wih[gidx], bih, gi, rows, 256, 768, 0);
    castb(h_state, t2, rows * 256);
    gemm(t2, wb_Whh[gidx], bhh, gh, rows, 256, 768, 0);
    gru_combine_kernel<<<cgrid(rows * 256), 256, 0, stream>>>(gi, gh, h_state, rows);
  };

  // ---- weights -> bf16 ----
  castb(lin1_W, wb_lin1, 256 * 64);
  cast_bf16_strided_kernel<<<cgrid(256 * 256), 256, 0, stream>>>(g_W1, 272, wb_gateWx, 256, 256);
  castb(g_W2, wb_gateW2, 256 * 256);
  const int gru_idx[5] = {11, 27, 31, 35, 43};
  for (int i = 0; i < 5; ++i) {
    castb(P(gru_idx[i] + 0), wb_Wih[i], 768 * 256);
    castb(P(gru_idx[i] + 1), wb_Whh[i], 768 * 256);
  }
  const int gat_idx[4] = {15, 19, 23, 39};
  for (int i = 0; i < 4; ++i) castb(P(gat_idx[i]), wb_gat[i], 256 * 256);
  castb(lin2_W, wb_lin2, 256 * 256);
  castb(head_W, wb_head, 256 * 256);

  // ---- x = leaky(x @ lin1_W^T + b) ----
  castb(x_in, XB, N * 64);
  gemm(XB, wb_lin1, lin1_b, X, N, 64, 256, 1);

  // ---- GATEConv ----
  castb(X, XB, N * 256);
  gemm(XB, wb_gateWx, nullptr, PX, N, 256, 256, 0);                 // px = x @ Wx^T
  rowdot_kernel<<<wgrid(N), 256, 0, stream>>>(X, g_attr, VR, N);    // xr
  gate_alpha_kernel<<<wgrid(E), 256, 0, stream>>>(PX, eattr, g_W1, g_attl, VR, row, col, AL, E);
  softmax_seg(AL, col, E, N);
  gemm(XB, wb_gateW2, nullptr, PX, N, 256, 256, 0);                 // msg = x @ W2^T
  rows_bias_kernel<<<cgrid(N * 256), 256, 0, stream>>>(HN, g_bias, N);
  scatter_wsum_kernel<<<wgrid(E), 256, 0, stream>>>(PX, EE, SS, row, col, HN, E);
  elu_kernel<<<cgrid(N * 256), 256, 0, stream>>>(HN, N * 256);
  run_gru(HN, X, 0, N, XB, XB2, GI, GH, P(13), P(14));

  // ---- GATConv atom layers ----
  for (int L = 0; L < 3; ++L) {
    const int gi_ = gat_idx[L];
    castb(X, XB, N * 256);
    gemm(XB, wb_gat[L], nullptr, PX, N, 256, 256, 0);               // xs
    rowdot_kernel<<<wgrid(N), 256, 0, stream>>>(PX, P(gi_ + 1), VR, N);
    rowdot_kernel<<<wgrid(N), 256, 0, stream>>>(PX, P(gi_ + 2), VR2, N);
    gat_alpha_kernel<<<cgrid(E), 256, 0, stream>>>(VR, VR2, row, col, AL, E);
    softmax_seg(AL, col, E, N);
    rows_bias_kernel<<<cgrid(N * 256), 256, 0, stream>>>(HN, P(gi_ + 3), N);
    scatter_wsum_kernel<<<wgrid(E), 256, 0, stream>>>(PX, EE, SS, row, col, HN, E);
    elu_kernel<<<cgrid(N * 256), 256, 0, stream>>>(HN, N * 256);
    run_gru(HN, X, 1 + L, N, XB, XB2, GI, GH, P(gru_idx[1 + L] + 2), P(gru_idx[1 + L] + 3));
  }

  // ---- molecule readout ----
  fill_kernel<<<cgrid(G * 256), 256, 0, stream>>>(OUTG, 0.f, G * 256);
  scatter_wsum_kernel<<<wgrid(N), 256, 0, stream>>>(X, nullptr, nullptr, nullptr, batch, OUTG, N);
  relu_kernel<<<cgrid(G * 256), 256, 0, stream>>>(OUTG, G * 256);
  castb(X, XB, N * 256);
  gemm(XB, wb_gat[3], nullptr, PX, N, 256, 256, 0);                 // xs (fixed)
  rowdot_kernel<<<wgrid(N), 256, 0, stream>>>(PX, P(40), VR, N);    // a_src
  for (int t = 0; t < 6; ++t) {
    castb(OUTG, OB, G * 256);
    gemm(OB, wb_gat[3], nullptr, HG, G, 256, 256, 0);               // xd
    rowdot_kernel<<<wgrid(G), 256, 0, stream>>>(HG, P(41), AD, G);  // xd @ att_dst
    mol_alpha_kernel<<<cgrid(N), 256, 0, stream>>>(VR, AD, batch, AL, N);
    softmax_seg(AL, batch, N, G);
    rows_bias_kernel<<<cgrid(G * 256), 256, 0, stream>>>(HG, P(42), G);
    scatter_wsum_kernel<<<wgrid(N), 256, 0, stream>>>(PX, EE, SS, nullptr, batch, HG, N);
    elu_kernel<<<cgrid(G * 256), 256, 0, stream>>>(HG, G * 256);
    run_gru(HG, OUTG, 4, G, OB, OB2, GIG, GHG, P(45), P(46));
  }

  // ---- dense head ----
  castb(OUTG, OB, G * 256);
  gemm(OB, wb_lin2, lin2_b, HG, G, 256, 256, 0);
  castb(HG, OB, G * 256);
  gemm(OB, wb_head, head_b, GIG, G, 256, 256, 0);
  head_kernel<<<wgrid(G), 256, 0, stream>>>(GIG, ln_g, ln_b, out_W, out_b, (float*)d_out, G);
}
